// VectorQuantizerEMA_45827301048596
// MI455X (gfx1250) — compile-verified, matrix operand order checked by reading, not run on hardware
//
#include <hip/hip_runtime.h>
#include <hip/hip_bf16.h>
#include <math.h>

typedef _Float16 half8_t  __attribute__((ext_vector_type(8)));
typedef _Float16 half16_t __attribute__((ext_vector_type(16)));
typedef float    float8_t __attribute__((ext_vector_type(8)));

#define K_CODES 1024
#define DIM     256
#define N_TOK   65536

// Async global->LDS copy of 32 bytes per lane (CDNA5, ASYNCcnt-tracked).
// INST_OFFSET is added to BOTH the LDS and the global address (ISA 08 §4.4).
#define ASYNC_TILE_LOAD(ldsoff, gptr)                                      \
    asm volatile("global_load_async_to_lds_b128 %0, %1, off\n\t"           \
                 "global_load_async_to_lds_b128 %0, %1, off offset:16"     \
                 :: "v"(ldsoff), "v"(gptr) : "memory")

#define WAIT_ASYNC0() asm volatile("s_wait_asynccnt 0x0" ::: "memory")

// ---------------- Kernel 0: prep (emb f32->f16, ||e||^2, zero accumulators) ----
__global__ __launch_bounds__(256) void vq_prep(const float* __restrict__ emb,
        _Float16* __restrict__ embh, float* __restrict__ eNorm,
        float* __restrict__ counts, float* __restrict__ lossAcc)
{
    const int k = blockIdx.x;     // one code row per block
    const int t = threadIdx.x;    // 256 threads == DIM
    float v = emb[(size_t)k * DIM + t];
    embh[(size_t)k * DIM + t] = (_Float16)v;
    float s = v * v;
    #pragma unroll
    for (int m = 16; m >= 1; m >>= 1) s += __shfl_xor(s, m, 32);
    __shared__ float red[8];
    if ((t & 31) == 0) red[t >> 5] = s;
    __syncthreads();
    if (t == 0) {
        float tot = 0.f;
        #pragma unroll
        for (int i = 0; i < 8; ++i) tot += red[i];
        eNorm[k]  = tot;
        counts[k] = 0.f;
        if (k == 0) *lossAcc = 0.f;
    }
}

// pack 16 f32 values, K = c0..c0+7 and c0+16..c0+23, into an f16 frag
__device__ __forceinline__ half16_t pack_a(const float* zr, int c0)
{
    float4 f0 = *reinterpret_cast<const float4*>(zr + c0);
    float4 f1 = *reinterpret_cast<const float4*>(zr + c0 + 4);
    float4 f2 = *reinterpret_cast<const float4*>(zr + c0 + 16);
    float4 f3 = *reinterpret_cast<const float4*>(zr + c0 + 20);
    half16_t h;
    h[0]=(_Float16)f0.x;  h[1]=(_Float16)f0.y;  h[2]=(_Float16)f0.z;  h[3]=(_Float16)f0.w;
    h[4]=(_Float16)f1.x;  h[5]=(_Float16)f1.y;  h[6]=(_Float16)f1.z;  h[7]=(_Float16)f1.w;
    h[8]=(_Float16)f2.x;  h[9]=(_Float16)f2.y;  h[10]=(_Float16)f2.z; h[11]=(_Float16)f2.w;
    h[12]=(_Float16)f3.x; h[13]=(_Float16)f3.y; h[14]=(_Float16)f3.z; h[15]=(_Float16)f3.w;
    return h;
}

// ---------------- Kernel 1: fused WMMA GEMM + running argmin -------------------
// grid 256 x 256 threads (8 waves). Wave w owns rows blk*256 + w*32 .. +31
// (two 16-row A tiles reusing each LDS B fragment twice).
__global__ __launch_bounds__(256) void vq_argmin(const float* __restrict__ z,
        const _Float16* __restrict__ embh, const float* __restrict__ eNorm,
        int* __restrict__ idxOut)
{
    __shared__ float eN[K_CODES];
    __shared__ __align__(16) _Float16 tile[2][16][264];  // double-buffered 16x256 tile

    const int tid  = threadIdx.x;
    const int wid  = tid >> 5;
    const int lane = tid & 31;
    const int col  = lane & 15;     // column slot within 16x16 C tile
    const int hi   = lane >> 4;     // K-stripe select (ISA A/B layout)
    const int rowBase = blockIdx.x * 256 + wid * 32;

    // stage ||e||^2 into LDS
    #pragma unroll
    for (int i = 0; i < 4; ++i) eN[tid + i * 256] = eNorm[tid + i * 256];

    // Two register-resident A slices (16 rows x 256 each), f32->f16.
    half16_t a[2][8];
    #pragma unroll
    for (int rt = 0; rt < 2; ++rt) {
        const float* zr = z + (size_t)(rowBase + rt * 16 + col) * DIM;
        #pragma unroll
        for (int ch = 0; ch < 8; ++ch)
            a[rt][ch] = pack_a(zr, ch * 32 + hi * 8);
    }

    float mv[2][8]; int mi[2][8];
    #pragma unroll
    for (int rt = 0; rt < 2; ++rt)
        #pragma unroll
        for (int r = 0; r < 8; ++r) { mv[rt][r] = INFINITY; mi[rt][r] = 0; }

    // per-thread async-copy addresses (32B per thread per tile)
    const int codeR = tid >> 4;   // 0..15 code row
    const int seg   = tid & 15;   // 16-half segment
    const _Float16* gsrc = embh + (size_t)codeR * DIM + seg * 16;   // advances 16*DIM/tile
    const unsigned ldsOff[2] = {
        (unsigned)(size_t)&tile[0][codeR][seg * 16],
        (unsigned)(size_t)&tile[1][codeR][seg * 16] };

    // prologue: async-prefetch tile 0 into buffer 0
    ASYNC_TILE_LOAD(ldsOff[0], gsrc);
    gsrc += 16 * DIM;
    WAIT_ASYNC0();
    __syncthreads();

    for (int t = 0; t < 64; ++t) {
        const int cur = t & 1;
        if (t < 63) {                       // overlap next tile fetch with compute
            ASYNC_TILE_LOAD(ldsOff[cur ^ 1], gsrc);
            gsrc += 16 * DIM;
        }

        // S(16x16) x2 = A(16x256) * B(256x16); each B fragment feeds 2 WMMAs
        float8_t acc0 = {0.f,0.f,0.f,0.f,0.f,0.f,0.f,0.f};
        float8_t acc1 = {0.f,0.f,0.f,0.f,0.f,0.f,0.f,0.f};
        const _Float16* tr = &tile[cur][col][0];
        #pragma unroll
        for (int ch = 0; ch < 8; ++ch) {
            const int c0 = ch * 32 + hi * 8;
            half8_t b0 = *reinterpret_cast<const half8_t*>(tr + c0);
            half8_t b1 = *reinterpret_cast<const half8_t*>(tr + c0 + 16);
            half16_t b = __builtin_shufflevector(b0, b1,
                0,1,2,3,4,5,6,7,8,9,10,11,12,13,14,15);
            acc0 = __builtin_amdgcn_wmma_f32_16x16x32_f16(
                false, a[0][ch], false, b, (short)0, acc0, false, false);
            acc1 = __builtin_amdgcn_wmma_f32_16x16x32_f16(
                false, a[1][ch], false, b, (short)0, acc1, false, false);
        }

        // dist = ||e||^2 - 2*(z.e)  (||z||^2 constant per row -> dropped)
        const int kcode = t * 16 + col;
        const float en = eN[kcode];
        #pragma unroll
        for (int r = 0; r < 8; ++r) {
            float d0 = __builtin_fmaf(-2.f, acc0[r], en);
            if (d0 < mv[0][r]) { mv[0][r] = d0; mi[0][r] = kcode; }
            float d1 = __builtin_fmaf(-2.f, acc1[r], en);
            if (d1 < mv[1][r]) { mv[1][r] = d1; mi[1][r] = kcode; }
        }

        WAIT_ASYNC0();       // own async portion of tile t+1 landed in LDS
        __syncthreads();     // everyone's portion landed; everyone done reading
    }

    // reduce argmin across the 16 column lanes (xor masks stay inside halves)
    #pragma unroll
    for (int rt = 0; rt < 2; ++rt) {
        #pragma unroll
        for (int r = 0; r < 8; ++r) {
            float v = mv[rt][r]; int i = mi[rt][r];
            #pragma unroll
            for (int msk = 1; msk <= 8; msk <<= 1) {
                float ov = __shfl_xor(v, msk, 32);
                int   oi = __shfl_xor(i, msk, 32);
                if (ov < v || (ov == v && oi < i)) { v = ov; i = oi; }
            }
            if (col == 0) idxOut[rowBase + rt * 16 + hi * 8 + r] = i;
        }
    }
}

// ---------------- Kernel 2: gather z_q, loss partials, histogram ---------------
// grid 2048 x 256 threads, 32 rows per block, float4 streams.
__global__ __launch_bounds__(256) void vq_gather(const float* __restrict__ z,
        const unsigned char* __restrict__ mask, const float* __restrict__ emb,
        const int* __restrict__ idx, float* __restrict__ out,
        float* __restrict__ counts, float* __restrict__ lossAcc)
{
    const int tid = threadIdx.x;
    const int rowBase = blockIdx.x * 32;
    const int sub = tid >> 6;   // 0..3 (row within group of 4)
    const int e4  = tid & 63;   // float4 slot within row
    float acc = 0.f;
    #pragma unroll
    for (int i = 0; i < 8; ++i) {
        const int row = rowBase + i * 4 + sub;
        const int k = idx[row];
        float4 q  = *reinterpret_cast<const float4*>(emb + (size_t)k   * DIM + e4 * 4);
        float4 zv = *reinterpret_cast<const float4*>(z   + (size_t)row * DIM + e4 * 4);
        *reinterpret_cast<float4*>(out + (size_t)row * DIM + e4 * 4) = q;  // STE fwd == z_q
        float dx = q.x - zv.x, dy = q.y - zv.y, dz = q.z - zv.z, dw = q.w - zv.w;
        acc += dx*dx + dy*dy + dz*dz + dw*dw;
    }
    #pragma unroll
    for (int msk = 16; msk >= 1; msk >>= 1) acc += __shfl_xor(acc, msk, 32);
    __shared__ float red[8];
    if ((tid & 31) == 0) red[tid >> 5] = acc;
    __syncthreads();
    if (tid == 0) {
        float tot = 0.f;
        #pragma unroll
        for (int i = 0; i < 8; ++i) tot += red[i];
        atomicAdd(lossAcc, tot);
    }
    if (tid < 32) {
        const int row = rowBase + tid;
        const float valid = mask[row] ? 0.f : 1.f;
        atomicAdd(&counts[idx[row]], valid);
    }
}

// ---------------- Kernel 3: finalize loss + perplexity -------------------------
__global__ __launch_bounds__(1024) void vq_finalize(const float* __restrict__ counts,
        const float* __restrict__ lossAcc, float* __restrict__ out)
{
    const int t = threadIdx.x;   // 1024 threads = 32 waves, one per code
    __shared__ float red[32];
    __shared__ float totalS;
    const float c = counts[t];
    float s = c;
    #pragma unroll
    for (int m = 16; m >= 1; m >>= 1) s += __shfl_xor(s, m, 32);
    if ((t & 31) == 0) red[t >> 5] = s;
    __syncthreads();
    if (t < 32) {
        float v = red[t];
        #pragma unroll
        for (int m = 16; m >= 1; m >>= 1) v += __shfl_xor(v, m, 32);
        if (t == 0) totalS = v;
    }
    __syncthreads();
    const float denom = fmaxf(totalS, 1.0f);
    const float p = c / denom;
    float term = p * logf(p + 1e-10f);
    __syncthreads();   // red reuse
    #pragma unroll
    for (int m = 16; m >= 1; m >>= 1) term += __shfl_xor(term, m, 32);
    if ((t & 31) == 0) red[t >> 5] = term;
    __syncthreads();
    if (t == 0) {
        float ent = 0.f;
        #pragma unroll
        for (int i = 0; i < 32; ++i) ent += red[i];
        out[(size_t)N_TOK * DIM]     = 0.25f * (*lossAcc) / ((float)N_TOK * (float)DIM);
        out[(size_t)N_TOK * DIM + 1] = expf(-ent);
    }
}

// ---------------- launch -------------------------------------------------------
extern "C" void kernel_launch(void* const* d_in, const int* in_sizes, int n_in,
                              void* d_out, int out_size, void* d_ws, size_t ws_size,
                              hipStream_t stream) {
    const float*         z    = (const float*)d_in[0];          // [64,1024,256] f32
    const unsigned char* mask = (const unsigned char*)d_in[1];  // [64,1024] bool
    const float*         emb  = (const float*)d_in[2];          // [1024,256] f32
    float* out = (float*)d_out;                                 // quantized | loss | ppl

    char* ws = (char*)d_ws;
    _Float16* embh    = (_Float16*)(ws);                 // 524288 B
    float*    eNorm   = (float*)(ws + 524288);           // 4096 B
    float*    counts  = (float*)(ws + 528384);           // 4096 B
    float*    lossAcc = (float*)(ws + 532480);           // 8 B (padded)
    int*      idxBuf  = (int*)(ws + 532488);             // 262144 B

    vq_prep    <<<K_CODES,     256, 0, stream>>>(emb, embh, eNorm, counts, lossAcc);
    vq_argmin  <<<N_TOK / 256, 256, 0, stream>>>(z, embh, eNorm, idxBuf);
    vq_gather  <<<N_TOK / 32,  256, 0, stream>>>(z, mask, emb, idxBuf, out, counts, lossAcc);
    vq_finalize<<<1,          1024, 0, stream>>>(counts, lossAcc, out);
}
